// S4_56788057588283
// MI455X (gfx1250) — matrix operand order repaired and checked by reading, not compiled, not run on hardware
//
#include <hip/hip_runtime.h>
#include <hip/hip_bf16.h>

// S4 forward for MI455X (gfx1250, wave32, WMMA).
// Shapes: H=512, N2=32, L=2048, BATCH=8, SEQ=2048.
//   d_in: x(8,2048,512) f32, log_dt(512), w_re(32), w_im(32),
//         B_re(512,2,32), B_im, C_re, C_im, D(512)   all f32
//   d_out: y(8,2048,512) f32
//   d_ws : k(512,2048) f32  (4 MB scratch)

#define S4_H     512
#define S4_N2    32
#define S4_L     2048
#define S4_B     8
#define S4_SEQ   2048
#define S4_TWOPI 6.283185307179586f

typedef __attribute__((ext_vector_type(16))) __bf16 v16bf;
typedef __attribute__((ext_vector_type(8)))  __bf16 v8bf;
typedef __attribute__((ext_vector_type(8)))  float  v8f;

// ---------------------------------------------------------------------------
// Kernel 1: build the SSM kernel k[h, 0..2047].
// One block per h.  Cauchy + Woodbury on the half spectrum, Hermitian extend,
// then inverse DFT factored as 2048 = 64 x 32 (two-step Cooley-Tukey) using
// an LDS twiddle table.
// ---------------------------------------------------------------------------
__global__ __launch_bounds__(256) void s4_build_kernel(
    const float* __restrict__ log_dt,
    const float* __restrict__ w_re, const float* __restrict__ w_im,
    const float* __restrict__ B_re, const float* __restrict__ B_im,
    const float* __restrict__ C_re, const float* __restrict__ C_im,
    float* __restrict__ kbuf)
{
    __shared__ float ctab[S4_L];        // cos(2*pi*m/L)
    __shared__ float stab[S4_L];        // sin(2*pi*m/L)
    __shared__ float Kre[S4_L];         // full Hermitian spectrum
    __shared__ float Kim[S4_L];
    __shared__ float Tre[64][33];       // inner-DFT partials (padded)
    __shared__ float Tim[64][33];
    __shared__ float vre[4][S4_N2];     // v[ij][n] = conj(C_i)*B_j
    __shared__ float vim[4][S4_N2];
    __shared__ float wsr[S4_N2];        // w*dt
    __shared__ float wsi[S4_N2];

    const int h   = blockIdx.x;
    const int tid = threadIdx.x;
    const float dt = __expf(log_dt[h]);

    // twiddle table
    for (int m = tid; m < S4_L; m += 256) {
        float ang = (S4_TWOPI / (float)S4_L) * (float)m;
        ctab[m] = __cosf(ang);
        stab[m] = __sinf(ang);
    }
    // v[ij][n] and scaled poles
    if (tid < 128) {
        const int n  = tid & 31;
        const int ij = tid >> 5;             // 0..3 = (i<<1)|j
        const int i  = ij >> 1, j = ij & 1;
        const float cr = C_re[h*64 + i*32 + n];
        const float ci = C_im[h*64 + i*32 + n];
        const float br = B_re[h*64 + j*32 + n];
        const float bi = B_im[h*64 + j*32 + n];
        vre[ij][n] = cr*br + ci*bi;          // conj(C)*B
        vim[ij][n] = cr*bi - ci*br;
    }
    if (tid < S4_N2) {
        wsr[tid] = w_re[tid] * dt;
        wsi[tid] = w_im[tid] * dt;
    }
    __syncthreads();

    // ---- Cauchy + Woodbury on f = 0..1024 ----
    for (int f = tid; f <= S4_L/2; f += 256) {
        float kfre, kfim;
        const float cf = ctab[f];
        const float sf = stab[f];
        if (f == S4_L/2) {
            // analytic limit at Nyquist: kf = dt * sum_n Re(v00[n])
            float s = 0.f;
            for (int n = 0; n < S4_N2; ++n) s += vre[0][n];
            kfre = dt * s; kfim = 0.f;
        } else {
            // freq = exp(-2*pi*i*f/L) = (cf, -sf);  z = 2(1-freq)/(1+freq)
            const float d   = (1.f + cf)*(1.f + cf) + sf*sf;   // |1+freq|^2
            const float zim = 4.f * sf / d;                    // z = i*zim
            float rre[4] = {0.f,0.f,0.f,0.f};
            float rim[4] = {0.f,0.f,0.f,0.f};
            #pragma unroll 4
            for (int n = 0; n < S4_N2; ++n) {
                const float dre  = -wsr[n];            // Re(z - w) = -Re(w*dt)
                const float d1im = zim - wsi[n];
                const float d2im = zim + wsi[n];
                const float q1 = __frcp_rn(dre*dre + d1im*d1im);
                const float q2 = __frcp_rn(dre*dre + d2im*d2im);
                const float i1re =  dre*q1, i1im = -d1im*q1;   // 1/(z-w)
                const float i2re =  dre*q2, i2im = -d2im*q2;   // 1/(z-conj(w))
                const float pre = i1re + i2re, pim = i1im + i2im;
                const float mre = i1re - i2re, dmi = i1im - i2im;
                #pragma unroll
                for (int ij = 0; ij < 4; ++ij) {
                    // v*inv1 + conj(v)*inv2
                    rre[ij] += vre[ij][n]*pre - vim[ij][n]*dmi;
                    rim[ij] += vre[ij][n]*pim + vim[ij][n]*mre;
                }
            }
            #pragma unroll
            for (int ij = 0; ij < 4; ++ij) { rre[ij] *= dt; rim[ij] *= dt; }
            // Woodbury: kf = r00 - r01*r10/(1+r11)
            const float tre = rre[1]*rre[2] - rim[1]*rim[2];
            const float tim = rre[1]*rim[2] + rim[1]*rre[2];
            const float wre = 1.f + rre[3], wim = rim[3];
            const float qd  = __frcp_rn(wre*wre + wim*wim);
            kfre = rre[0] - (tre*wre + tim*wim)*qd;
            kfim = rim[0] - (tim*wre - tre*wim)*qd;
            // *= 2/(1+freq) = (2/d)*(1+cf, sf)
            const float g   = 2.f / d;
            const float cre = (1.f + cf)*g, cim = sf*g;
            const float nre = kfre*cre - kfim*cim;
            const float nim = kfre*cim + kfim*cre;
            kfre = nre; kfim = nim;
        }
        Kre[f] = kfre; Kim[f] = kfim;
        if (f >= 1 && f <= S4_L/2 - 1) {       // Hermitian extension
            Kre[S4_L - f] =  kfre;
            Kim[S4_L - f] = -kfim;
        }
    }
    __syncthreads();

    // ---- inverse DFT, step 1: T[f0][l mod 32] = sum_f1 K[f0+64*f1] e^{2pi i f1 r/32}
    for (int e = tid; e < 64*32; e += 256) {
        const int f0 = e >> 5, rr = e & 31;
        float are = 0.f, aim = 0.f;
        #pragma unroll 4
        for (int f1 = 0; f1 < 32; ++f1) {
            const int fi  = f0 + (f1 << 6);
            const int idx = ((f1 * rr) << 6) & (S4_L - 1);
            const float c = ctab[idx], s = stab[idx];
            const float kr = Kre[fi], ki = Kim[fi];
            are += kr*c - ki*s;
            aim += kr*s + ki*c;
        }
        Tre[f0][rr] = are; Tim[f0][rr] = aim;
    }
    __syncthreads();

    // ---- step 2: k[l] = (1/L) Re( sum_f0 T[f0][l%32] e^{2pi i f0 l / L} )
    for (int l = tid; l < S4_L; l += 256) {
        const int rr = l & 31;
        float acc = 0.f;
        #pragma unroll 8
        for (int f0 = 0; f0 < 64; ++f0) {
            const int idx = (f0 * l) & (S4_L - 1);
            acc += Tre[f0][rr]*ctab[idx] - Tim[f0][rr]*stab[idx];
        }
        kbuf[h*S4_L + l] = acc * (1.f / (float)S4_L);
    }
}

// ---------------------------------------------------------------------------
// Kernel 2: causal convolution via WMMA Toeplitz matmuls.
// One block per h (256 threads = 8 waves).  Whole x[:,:,h] slice cached in
// LDS as bf16 (row 8 zeroed so N-lanes 8..15 feed zeros to WMMA), kernel k
// cached with a 32-entry zero pad for out-of-range (future) taps.
// Each wave owns output-time tiles of 16; per 32-wide tau chunk:
//   A[m,kk] = k[t0+m - (s0+kk)]  (CDNA5 16-bit A layout, gathered)
//   B[kk,n] = x[n, s0+kk]        (two ds_load_b128 per lane)
//   acc = v_wmma_f32_16x16x32_bf16(A, B, acc)
// Chunk loop unrolled 2x so DS gathers of chunk c+1 co-execute with the
// in-flight (TRANS-tracked) WMMA of chunk c.
// ---------------------------------------------------------------------------
__global__ __launch_bounds__(256) void s4_conv_kernel(
    const float* __restrict__ x,
    const float* __restrict__ kbuf,
    const float* __restrict__ Dvec,
    float* __restrict__ out)
{
    __shared__ __bf16 kh[32 + S4_L];        // [0..31] = zero pad
    __shared__ __bf16 xh[9 * S4_SEQ];       // rows 0..7 = batches, row 8 = zeros

    const int h   = blockIdx.x;
    const int tid = threadIdx.x;

    if (tid < 32) kh[tid] = (__bf16)0.f;
    for (int l = tid; l < S4_L; l += 256)
        kh[32 + l] = (__bf16)kbuf[h*S4_L + l];
    for (int e = tid; e < S4_B * S4_SEQ; e += 256) {
        const int b = e >> 11, s = e & (S4_SEQ - 1);
        xh[e] = (__bf16)x[((size_t)(b*S4_SEQ + s))*S4_H + h];
    }
    for (int s = tid; s < S4_SEQ; s += 256)
        xh[8*S4_SEQ + s] = (__bf16)0.f;
    __syncthreads();

    const float Dh   = Dvec[h];
    const int   wid  = tid >> 5;
    const int   lane = tid & 31;
    const int   m    = lane & 15;          // A row / D column group
    const int   hi   = lane >> 4;          // lane half
    const int   kbA  = hi * 8;             // A K-base for this lane half
    const int   kbB  = hi * 8;             // B K-base for this lane half
    const int   n    = lane & 15;          // D column = batch (8..15 unused)
    const int   nr   = (n < 8) ? n : 8;    // row 8 = zeros
    const __bf16* xrow = &xh[nr * S4_SEQ];

    for (int tile = wid; tile < S4_SEQ / 16; tile += 8) {
        const int t0  = tile << 4;
        const int nch = (tile >> 1) + 1;   // causal: s0 <= t0
        v8f acc = {};
        #pragma unroll 2
        for (int c = 0; c < nch; ++c) {
            const int s0 = c << 5;
            // A: Toeplitz kernel tile (per CDNA5 16-bit A-matrix layout)
            v16bf A;
            const int base = 32 + t0 + m - s0 - kbA;   // >= 1 always
            #pragma unroll
            for (int j = 0; j < 8; ++j) A[j]     = kh[base - j];
            #pragma unroll
            for (int j = 0; j < 8; ++j) A[8 + j] = kh[base - 16 - j];
            // B: two 8-sample runs (k = 8h+s, 16+8h+s) -> two b128 LDS loads
            const v8bf blo = *(const v8bf*)(xrow + s0 + kbB);
            const v8bf bhi = *(const v8bf*)(xrow + s0 + kbB + 16);
            const v16bf Bv = __builtin_shufflevector(
                blo, bhi, 0,1,2,3,4,5,6,7,8,9,10,11,12,13,14,15);
            acc = __builtin_amdgcn_wmma_f32_16x16x32_bf16(
                false, A, false, Bv, (short)0, acc, false, false);
        }
        // store + D*x skip term (f32 x straight from global for precision)
        if (n < 8) {
            #pragma unroll
            for (int j = 0; j < 8; ++j) {
                const int t = t0 + j + hi*8;
                const size_t off = ((size_t)(n*S4_SEQ + t))*S4_H + h;
                out[off] = acc[j] + Dh * x[off];
            }
        }
    }
}

extern "C" void kernel_launch(void* const* d_in, const int* in_sizes, int n_in,
                              void* d_out, int out_size, void* d_ws, size_t ws_size,
                              hipStream_t stream) {
    const float* x      = (const float*)d_in[0];
    const float* log_dt = (const float*)d_in[1];
    const float* w_re   = (const float*)d_in[2];
    const float* w_im   = (const float*)d_in[3];
    const float* B_re   = (const float*)d_in[4];
    const float* B_im   = (const float*)d_in[5];
    const float* C_re   = (const float*)d_in[6];
    const float* C_im   = (const float*)d_in[7];
    const float* Dv     = (const float*)d_in[8];
    float* out  = (float*)d_out;
    float* kbuf = (float*)d_ws;            // 512*2048 f32 = 4 MB scratch

    s4_build_kernel<<<S4_H, 256, 0, stream>>>(log_dt, w_re, w_im,
                                              B_re, B_im, C_re, C_im, kbuf);
    s4_conv_kernel<<<S4_H, 256, 0, stream>>>(x, kbuf, Dv, out);
}
